// FlashAttention_10316511445594
// MI455X (gfx1250) — matrix operand order repaired and checked by reading, not compiled, not run on hardware
//
#include <hip/hip_runtime.h>

typedef __attribute__((ext_vector_type(16))) __bf16 v16bf;
typedef __attribute__((ext_vector_type(8)))  float  v8f;

#define B_      4
#define L_      2048
#define H_      16
#define D_      64
#define BLK     32
#define QTILE   256            // per block: 8 waves x 2 tiles x 16 rows
#define TC      (L_ / BLK)
#define NEG_INF (-1e10f)
#define EPS_    (1e-10f)

static __device__ __forceinline__ unsigned short f2bf(float f) {
  union { float f; unsigned u; } x; x.f = f;
  unsigned u = x.u;
  unsigned r = 0x7FFFu + ((u >> 16) & 1u);   // round to nearest even
  return (unsigned short)((u + r) >> 16);
}
static __device__ __forceinline__ __bf16 us2bf(unsigned short s) {
  union { unsigned short s; __bf16 b; } x; x.s = s; return x.b;
}
static __device__ __forceinline__ v8f wmma_bf16(v16bf a, v16bf b, v8f c) {
  return __builtin_amdgcn_wmma_f32_16x16x32_bf16(false, a, false, b, (short)0, c, false, false);
}

// FlashAttention forward. One (b,h,256-row q tile) per 256-thread block (8 waves),
// two 16-row q tiles per wave (K/V fragments reused for both -> 16 WMMA / KV block).
// Transposed-S formulation: S^T = K*Q^T puts q on the lane, so softmax stats are
// per-lane scalars and P's C-layout equals the A-fragment layout for P@V.
// Unnormalized accumulator U = l*O (telescoped reference recurrence), U/l at the end;
// exp(mb-m_new) folded into the exponent: P_acc = exp(S - m_new).
__global__ __launch_bounds__(256) void FlashAttention_10316511445594_kernel(
    const float* __restrict__ Qg, const float* __restrict__ Kg,
    const float* __restrict__ Vg, float* __restrict__ Og) {
  // K block row-major (32 x 64 bf16, rows padded to 33 dwords: conflict-free gathers),
  // V block transposed (64 x 32 bf16, rows padded to 34 halves).
  __shared__ unsigned int   sK[32 * 33];
  __shared__ unsigned short sV[64 * 34];

  const int tid  = threadIdx.x;
  const int wave = tid >> 5, lane = tid & 31;
  const int g = lane >> 4, n = lane & 15;

  const int qtiles = L_ / QTILE;
  const int qt = blockIdx.x % qtiles;
  const int bh = blockIdx.x / qtiles;
  const int hh = bh % H_, b = bh / H_;
  const int qrowA = qt * QTILE + wave * 16;   // q tile A
  const int qrowB = qrowA + 128;              // q tile B

  const float scale = 0.125f;                 // 1/sqrt(64)

  // ---- Q tiles -> 32x16 bf16 B-fragments (for S^T = K * Q^T), pre-scaled ----
  // B fragment in contiguous k order (K fragments match): lane N(q) = n, VGPR r half h -> K(d) = 32f + 16g + 2r + h
  v16bf qbA[2], qbB[2];
  {
    const long qoffA = (((long)b * L_ + (qrowA + n)) * H_ + hh) * D_;
    const long qoffB = (((long)b * L_ + (qrowB + n)) * H_ + hh) * D_;
#pragma unroll
    for (int f = 0; f < 2; ++f)
#pragma unroll
      for (int r = 0; r < 8; ++r) {
        int d0 = 32 * f + 16 * g + 2 * r;
        float2 qa = *(const float2*)(Qg + qoffA + d0);
        float2 qv = *(const float2*)(Qg + qoffB + d0);
        qbA[f][2 * r]     = (__bf16)(qa.x * scale);
        qbA[f][2 * r + 1] = (__bf16)(qa.y * scale);
        qbB[f][2 * r]     = (__bf16)(qv.x * scale);
        qbB[f][2 * r + 1] = (__bf16)(qv.y * scale);
      }
  }

  v8f oA[4] = {}, oB[4] = {};   // U = l*O per tile (row q = 8g+r, col d = n)
  float mA = NEG_INF, lA = 0.f, mB = NEG_INF, lB = 0.f;

  const int kvs = tid >> 3;     // staging: kv row 0..31
  const int d0s = (tid & 7) * 8;

  for (int j = 0; j < TC; ++j) {
    // ---- stage K (row-major) and V (transposed) blocks into LDS as bf16 ----
    {
      const long off = (((long)b * L_ + (j * BLK + kvs)) * H_ + hh) * D_ + d0s;
      float4 k0 = *(const float4*)(Kg + off);
      float4 k1 = *(const float4*)(Kg + off + 4);
      unsigned int* kd = &sK[kvs * 33 + (d0s >> 1)];
      kd[0] = (unsigned)f2bf(k0.x) | ((unsigned)f2bf(k0.y) << 16);
      kd[1] = (unsigned)f2bf(k0.z) | ((unsigned)f2bf(k0.w) << 16);
      kd[2] = (unsigned)f2bf(k1.x) | ((unsigned)f2bf(k1.y) << 16);
      kd[3] = (unsigned)f2bf(k1.z) | ((unsigned)f2bf(k1.w) << 16);

      float4 v0 = *(const float4*)(Vg + off);
      float4 v1 = *(const float4*)(Vg + off + 4);
      float vv[8] = {v0.x, v0.y, v0.z, v0.w, v1.x, v1.y, v1.z, v1.w};
#pragma unroll
      for (int i = 0; i < 8; ++i) sV[(d0s + i) * 34 + kvs] = f2bf(vv[i]);

      if (j + 1 < TC) {  // next-block prefetch -> global_prefetch_b8
        const long noff = (((long)b * L_ + ((j + 1) * BLK + kvs)) * H_ + hh) * D_ + d0s;
        __builtin_prefetch(Kg + noff, 0, 0);
        __builtin_prefetch(Vg + noff, 0, 0);
      }
    }
    __syncthreads();

    // ---- S^T = K * Q^T for both q tiles; K fragments loaded once ----
    // A layout: lane M(kv) = 16t + n, VGPR r half h -> dword = 16f + 8g + r
    v8f sA[2], sB[2];
#pragma unroll
    for (int t = 0; t < 2; ++t) {
      v16bf ka0, ka1;
#pragma unroll
      for (int r = 0; r < 8; ++r) {
        int di = 8 * g + r;
        unsigned w0 = sK[(16 * t + n) * 33 + di];
        unsigned w1 = sK[(16 * t + n) * 33 + 16 + di];
        ka0[2 * r] = us2bf((unsigned short)w0); ka0[2 * r + 1] = us2bf((unsigned short)(w0 >> 16));
        ka1[2 * r] = us2bf((unsigned short)w1); ka1[2 * r + 1] = us2bf((unsigned short)(w1 >> 16));
      }
      v8f accA = {}, accB = {};
      accA = wmma_bf16(ka0, qbA[0], accA);
      accA = wmma_bf16(ka1, qbA[1], accA);
      accB = wmma_bf16(ka0, qbB[0], accB);
      accB = wmma_bf16(ka1, qbB[1], accB);
      sA[t] = accA; sB[t] = accB;
    }
    // Lane holds S[q = n][kv = 16t + 8g + r] in s*[t][r]: one q row per lane per tile.

    // ---- online softmax (per-lane scalars), exp(mb - m_new) folded into exponent ----
    v16bf paA, paB;
    float eoA, eoB;
    {
      float mb = NEG_INF;
#pragma unroll
      for (int t = 0; t < 2; ++t)
#pragma unroll
        for (int r = 0; r < 8; ++r) mb = fmaxf(mb, sA[t][r]);
      mb = fmaxf(mb, __shfl_xor(mb, 16, 32));
      float mn = fmaxf(mA, mb);
      eoA = __expf(mA - mn);
      float ebp = __expf(mb - mn);
      float p[16], lb = 0.f;
#pragma unroll
      for (int i = 0; i < 16; ++i) { p[i] = __expf(sA[i >> 3][i & 7] - mn); lb += p[i]; }
      lb += __shfl_xor(lb, 16, 32);
      lA = eoA * lA + lb + ebp * EPS_;   // == eo*l + eb*(sum exp(S-mb) + eps)
      mA = mn;
#pragma unroll
      for (int i = 0; i < 16; ++i) paA[i] = (__bf16)p[i];
    }
    {
      float mb = NEG_INF;
#pragma unroll
      for (int t = 0; t < 2; ++t)
#pragma unroll
        for (int r = 0; r < 8; ++r) mb = fmaxf(mb, sB[t][r]);
      mb = fmaxf(mb, __shfl_xor(mb, 16, 32));
      float mn = fmaxf(mB, mb);
      eoB = __expf(mB - mn);
      float ebp = __expf(mb - mn);
      float p[16], lb = 0.f;
#pragma unroll
      for (int i = 0; i < 16; ++i) { p[i] = __expf(sB[i >> 3][i & 7] - mn); lb += p[i]; }
      lb += __shfl_xor(lb, 16, 32);
      lB = eoB * lB + lb + ebp * EPS_;
      mB = mn;
#pragma unroll
      for (int i = 0; i < 16; ++i) paB[i] = (__bf16)p[i];
    }

    // ---- rescale U rows by eo (row q = 8g + r lives in lane 8g + r) ----
    {
      float ea[8], eb2[8];
#pragma unroll
      for (int r = 0; r < 8; ++r) {
        ea[r]  = __shfl(eoA, 8 * g + r, 32);
        eb2[r] = __shfl(eoB, 8 * g + r, 32);
      }
#pragma unroll
      for (int t = 0; t < 4; ++t)
#pragma unroll
        for (int r = 0; r < 8; ++r) { oA[t][r] *= ea[r]; oB[t][r] *= eb2[r]; }
    }

    // ---- U += P @ V for both tiles; V fragments loaded once ----
    // B layout for V (32x16): lane N(d) = 16t + n, VGPR r half h -> K(kv) = (r<4?0:8) + 8g + 2r + h
#pragma unroll
    for (int t = 0; t < 4; ++t) {
      v16bf vb;
#pragma unroll
      for (int r = 0; r < 8; ++r) {
        unsigned w = *(const unsigned*)&sV[(16 * t + n) * 34 + ((r < 4) ? 0 : 8) + 8 * g + 2 * r];
        vb[2 * r] = us2bf((unsigned short)w); vb[2 * r + 1] = us2bf((unsigned short)(w >> 16));
      }
      oA[t] = wmma_bf16(paA, vb, oA[t]);
      oB[t] = wmma_bf16(paB, vb, oB[t]);
    }
    __syncthreads();
  }

  // ---- O = U / l ; write back in [B, L, H, D] ----
  float lrA[8], lrB[8];
#pragma unroll
  for (int r = 0; r < 8; ++r) {
    lrA[r] = __shfl(lA, 8 * g + r, 32);
    lrB[r] = __shfl(lB, 8 * g + r, 32);
  }
#pragma unroll
  for (int t = 0; t < 4; ++t)
#pragma unroll
    for (int r = 0; r < 8; ++r) {
      int d = 16 * t + n;
      int rowA = qrowA + 8 * g + r;
      int rowB = qrowB + 8 * g + r;
      Og[(((long)b * L_ + rowA) * H_ + hh) * D_ + d] = oA[t][r] / lrA[r];
      Og[(((long)b * L_ + rowB) * H_ + hh) * D_ + d] = oB[t][r] / lrB[r];
    }
}

extern "C" void kernel_launch(void* const* d_in, const int* in_sizes, int n_in,
                              void* d_out, int out_size, void* d_ws, size_t ws_size,
                              hipStream_t stream) {
  (void)in_sizes; (void)n_in; (void)out_size; (void)d_ws; (void)ws_size;
  const float* Q = (const float*)d_in[0];
  const float* K = (const float*)d_in[1];
  const float* V = (const float*)d_in[2];
  float* O = (float*)d_out;
  dim3 grid(B_ * H_ * (L_ / QTILE));   // 512 blocks
  dim3 block(256);                      // 8 wave32 waves
  FlashAttention_10316511445594_kernel<<<grid, block, 0, stream>>>(Q, K, V, O);
}
